// RNABodyDeepModel_25769804367
// MI455X (gfx1250) — matrix operand order repaired and checked by reading, not compiled, not run on hardware
//
#include <hip/hip_runtime.h>
#include <hip/hip_bf16.h>

// ---------------------------------------------------------------------------
// CDNA5 (gfx1250) implementation. wave32, WMMA f32_16x16x32_f16 everywhere a
// GEMM appears (dense layers, conv-as-gemm, GNN aggregation, attention, RNN
// recurrent matmuls). All activations fp32 in memory; tiles converted to f16
// while staging into LDS, accumulation in fp32. Feature-concat copies go
// through the CDNA5 async LDS path (GLOBAL_*_ASYNC_*_LDS_B128 + ASYNCcnt).
// ---------------------------------------------------------------------------

typedef __attribute__((ext_vector_type(16))) _Float16 v16h;
typedef __attribute__((ext_vector_type(8)))  _Float16 v8h;
typedef __attribute__((ext_vector_type(8)))  float    v8f;

#define WMMA(a, b, c) __builtin_amdgcn_wmma_f32_16x16x32_f16(false, (a), false, (b), (short)0, (c), false, false)

__device__ __forceinline__ float sigf(float x) { return 1.0f / (1.0f + expf(-x)); }

__device__ __forceinline__ float act_f(float x, int act) {
  if (act == 1) return fmaxf(x, 0.0f);
  if (act == 2) {  // gelu tanh approx
    float c = x + 0.044715f * x * x * x;
    return 0.5f * x * (1.0f + tanhf(0.7978845608028654f * c));
  }
  return x;
}

// A-fragment (16xK tile, f16): lane m = lane&15, hi = lane>>4.
// element j -> k = hi*16 + j  (contiguous 16 halves)
__device__ __forceinline__ v16h frag_a(const _Float16* base, int stride, int m0, int k0, int lane) {
  int m  = m0 + (lane & 15);
  int hi = (lane >> 4) & 1;
  const _Float16* r = base + (size_t)m * stride + k0;
  v8h x0 = *(const v8h*)(r + hi * 16);
  v8h x1 = *(const v8h*)(r + hi * 16 + 8);
  v16h out;
#pragma unroll
  for (int j = 0; j < 8; ++j) { out[j] = x0[j]; out[j + 8] = x1[j]; }
  return out;
}

// B-fragment (Kx16 tile, f16), LDS stored [n][k]: lane n = lane&15,
// element j -> k = hi*16 + j  (contiguous 16 halves)
__device__ __forceinline__ v16h frag_b(const _Float16* base, int stride, int n0, int k0, int lane) {
  int n  = n0 + (lane & 15);
  int hi = (lane >> 4) & 1;
  return *(const v16h*)(base + (size_t)n * stride + k0 + hi * 16);
}

__device__ __forceinline__ void store_tile(float* C, int ldc, const v8f& acc, int gm0, int gn0,
                                           int M, int N, const float* bias, int act, int lane) {
  int n = gn0 + (lane & 15);
  if (n >= N) return;
  float bv = bias ? bias[n] : 0.0f;
  int hi = lane >> 4;
#pragma unroll
  for (int j = 0; j < 8; ++j) {
    int m = gm0 + j + 8 * hi;
    if (m < M) C[(size_t)m * ldc + n] = act_f(acc[j] + bv, act);
  }
}

// ---------------------------------------------------------------------------
// Generic GEMM: C[M,N] = act(A[M,K] * B[K,N] + bias), fp32 in/out, f16 WMMA.
// Optional batch (grid.z) with per-operand strides. Optional conv mode:
// logical A(row=(b*len+t), col=tap*cin+c) = X[b*len + t + tap - pad][c],
// zero-padded (lax 'SAME').  Block = 64x64 tile, 256 threads = 8 waves,
// each wave = 16x32 (two 16x16x32 WMMA accumulators).
// ---------------------------------------------------------------------------
__global__ __launch_bounds__(256) void k_gemm(
    const float* __restrict__ A, int lda, long long sA,
    const float* __restrict__ B, int ldb, long long sB,
    const float* __restrict__ bias,
    float* __restrict__ C, int ldc, long long sC,
    int M, int N, int K, int act,
    int conv_ksz, int conv_cin, int conv_pad, int conv_len) {
  __shared__ __align__(64) _Float16 As[64 * 32];
  __shared__ __align__(64) _Float16 Bs[64 * 32];

  A += (size_t)blockIdx.z * sA;
  B += (size_t)blockIdx.z * sB;
  C += (size_t)blockIdx.z * sC;

  const int tid   = threadIdx.x;
  const int lane  = tid & 31;
  const int wave  = tid >> 5;
  const int wm    = wave & 3;    // 4 m-subtiles of 16
  const int wn    = wave >> 2;   // 2 n-subtiles of 32
  const int m_blk = blockIdx.y * 64;
  const int n_blk = blockIdx.x * 64;

  v8f acc0{}; v8f acc1{};
  const int nK = (K + 31) >> 5;

  for (int kt = 0; kt < nK; ++kt) {
    const int k0 = kt * 32;
    // ---- stage A tile (64x32) ----
#pragma unroll
    for (int it = 0; it < 8; ++it) {
      int idx = tid + it * 256;
      int mm = idx >> 5, kk = idx & 31;
      int gm = m_blk + mm, gk = k0 + kk;
      float v = 0.0f;
      if (gm < M && gk < K) {
        if (conv_ksz > 0) {
          int b = gm / conv_len, t = gm % conv_len;
          int tap = gk / conv_cin, c = gk - tap * conv_cin;
          int ts = t + tap - conv_pad;
          if (ts >= 0 && ts < conv_len) v = A[(size_t)(b * conv_len + ts) * lda + c];
        } else {
          v = A[(size_t)gm * lda + gk];
          if (gk + 32 < K) __builtin_prefetch(&A[(size_t)gm * lda + gk + 32]);
        }
      }
      As[mm * 32 + kk] = (_Float16)v;
    }
    // ---- stage B tile (32x64), stored transposed [n][k] ----
#pragma unroll
    for (int it = 0; it < 8; ++it) {
      int idx = tid + it * 256;
      int nn = idx & 63, kk = idx >> 6;
      int gn = n_blk + nn, gk = k0 + kk;
      float v = 0.0f;
      if (gn < N && gk < K) {
        v = B[(size_t)gk * ldb + gn];
        if (gk + 32 < K) __builtin_prefetch(&B[(size_t)(gk + 32) * ldb + gn]);
      }
      Bs[nn * 32 + kk] = (_Float16)v;
    }
    __syncthreads();

    v16h a  = frag_a(As, 32, wm * 16, 0, lane);
    v16h b0 = frag_b(Bs, 32, wn * 32, 0, lane);
    v16h b1 = frag_b(Bs, 32, wn * 32 + 16, 0, lane);
    acc0 = WMMA(a, b0, acc0);
    acc1 = WMMA(a, b1, acc1);
    __syncthreads();
  }

  store_tile(C, ldc, acc0, m_blk + wm * 16, n_blk + wn * 32, M, N, bias, act, lane);
  store_tile(C, ldc, acc1, m_blk + wm * 16, n_blk + wn * 32 + 16, M, N, bias, act, lane);
}

// ---------------------------------------------------------------------------
// LayerNorm over last dim D (<=128). One wave per row, shuffle reductions.
// ---------------------------------------------------------------------------
__global__ __launch_bounds__(256) void k_ln(const float* __restrict__ x, int ldx,
                                            float* __restrict__ y, int ldy,
                                            const float* __restrict__ gg,
                                            const float* __restrict__ bb,
                                            int rows, int D) {
  int wave = threadIdx.x >> 5, lane = threadIdx.x & 31;
  int row = blockIdx.x * 8 + wave;
  if (row >= rows) return;
  const float* xr = x + (size_t)row * ldx;
  float s = 0.0f, s2 = 0.0f;
  for (int j = lane; j < D; j += 32) { float v = xr[j]; s += v; s2 += v * v; }
#pragma unroll
  for (int m = 16; m > 0; m >>= 1) { s += __shfl_xor(s, m, 32); s2 += __shfl_xor(s2, m, 32); }
  float mean = s / (float)D;
  float var  = s2 / (float)D - mean * mean;
  float inv  = rsqrtf(var + 0.001f);
  float* yr = y + (size_t)row * ldy;
  for (int j = lane; j < D; j += 32) yr[j] = (xr[j] - mean) * inv * gg[j] + bb[j];
}

// ---------------------------------------------------------------------------
// Strided column-block copy for feature concatenation, routed through LDS via
// CDNA5 async LDS transfers (B128, ASYNCcnt-tracked). Requires cols % 4 == 0
// and 16B-aligned row bases (true for every concat in this model: all leading
// dims are multiples of 4 floats, workspace is 256B aligned).
// Generic->LDS address: ISA flat-aperture rule, LDS address is the low 32 bits of the generic one.
// ---------------------------------------------------------------------------
__global__ __launch_bounds__(256) void k_copy_async(const float* __restrict__ src, int lds_,
                                                    float* __restrict__ dst, int ldd,
                                                    int rows, int cols4) {
  __shared__ __align__(16) float stage[256 * 4];
  const int tid = threadIdx.x;
  unsigned lds_slot = (unsigned)(size_t)&stage[tid * 4];
  size_t total = (size_t)rows * cols4;
  for (size_t i = (size_t)blockIdx.x * blockDim.x + tid; i < total;
       i += (size_t)gridDim.x * blockDim.x) {
    size_t r = i / cols4;
    int c = (int)(i - r * cols4) * 4;
    unsigned long long gsrc = (unsigned long long)(size_t)(src + r * lds_ + c);
    unsigned long long gdst = (unsigned long long)(size_t)(dst + r * ldd + c);
    asm volatile("global_load_async_to_lds_b128 %0, %1, off"
                 :: "v"(lds_slot), "v"(gsrc) : "memory");
    asm volatile("s_wait_asynccnt 0x0" ::: "memory");
    asm volatile("global_store_async_from_lds_b128 %0, %1, off"
                 :: "v"(gdst), "v"(lds_slot) : "memory");
    asm volatile("s_wait_asynccnt 0x0" ::: "memory");
  }
}

// ---------------------------------------------------------------------------
// edges = concat([edge_inputs, relu(edge_inputs @ edgeW)]) stored as 6
// contiguous [8][512][512] planes so each adjacency slice is a dense GEMM A.
// ---------------------------------------------------------------------------
__global__ __launch_bounds__(256) void k_edges(const float* __restrict__ e,
                                               const float* __restrict__ w,
                                               float* __restrict__ out) {
  const size_t P = 8ull * 512 * 512;
  for (size_t i = (size_t)blockIdx.x * blockDim.x + threadIdx.x; i < P;
       i += (size_t)gridDim.x * blockDim.x) {
    const float* src = e + i * 5;
    float acc = 0.0f;
#pragma unroll
    for (int c = 0; c < 5; ++c) { float v = src[c]; out[c * P + i] = v; acc += v * w[c]; }
    out[5 * P + i] = fmaxf(acc, 0.0f);
  }
}

// ---------------------------------------------------------------------------
// Fused MHA core: O = softmax(Q K^T / sqrt(32)) V for (b, h, 16-query tile).
// One wave per block; 16x512 f32 score strip in LDS; WMMA for both matmuls.
// Q/K/V/O layout: [b][t][h*32 + d] (ld = 64).
// ---------------------------------------------------------------------------
__global__ __launch_bounds__(32) void k_attn(const float* __restrict__ Q,
                                             const float* __restrict__ K,
                                             const float* __restrict__ V,
                                             float* __restrict__ O) {
  __shared__ float sc[16][512];
  const int lane = threadIdx.x & 31;
  const int hi = lane >> 4, ln = lane & 15;
  const int qt = blockIdx.x;        // 0..31
  const int bh = blockIdx.y;        // 0..15
  const int b = bh >> 1, h = bh & 1;
  const int t0 = qt * 16;
  const size_t ld = 64;
  const float* Qb = Q + (size_t)(b * 512) * ld + h * 32;
  const float* Kb = K + (size_t)(b * 512) * ld + h * 32;
  const float* Vb = V + (size_t)(b * 512) * ld + h * 32;

  // Q fragment: 16 rows x 32 kdim (single K-step)
  v16h aq;
  {
    const float* r = Qb + (size_t)(t0 + ln) * ld;
#pragma unroll
    for (int j = 0; j < 16; ++j) {
      int kd = hi * 16 + j;
      aq[j] = (_Float16)r[kd];
    }
  }
  const float scl = 0.17677669529663687f;  // 1/sqrt(32)
  for (int kt = 0; kt < 32; ++kt) {        // 32 key tiles of 16
    v16h bk;
    const float* kr = Kb + (size_t)(kt * 16 + ln) * ld;
#pragma unroll
    for (int j = 0; j < 16; ++j) bk[j] = (_Float16)kr[hi * 16 + j];
    v8f acc{};
    acc = WMMA(aq, bk, acc);
#pragma unroll
    for (int j = 0; j < 8; ++j) sc[j + 8 * hi][kt * 16 + ln] = acc[j] * scl;
  }
  // softmax rows (single wave; LDS same-wave ops are in-order)
  for (int r = 0; r < 16; ++r) {
    float mx = -1e30f;
    for (int j = lane; j < 512; j += 32) mx = fmaxf(mx, sc[r][j]);
#pragma unroll
    for (int m = 16; m > 0; m >>= 1) mx = fmaxf(mx, __shfl_xor(mx, m, 32));
    float sm = 0.0f;
    for (int j = lane; j < 512; j += 32) { float e = expf(sc[r][j] - mx); sc[r][j] = e; sm += e; }
#pragma unroll
    for (int m = 16; m > 0; m >>= 1) sm += __shfl_xor(sm, m, 32);
    float inv = 1.0f / sm;
    for (int j = lane; j < 512; j += 32) sc[r][j] *= inv;
  }
  // O = P(16x512) @ V(512x32): 16 K-chunks of 32 keys
  v8f o0{}, o1{};
  for (int kt = 0; kt < 16; ++kt) {
    v16h ap;
#pragma unroll
    for (int j = 0; j < 16; ++j) {
      int kk = hi * 16 + j;
      ap[j] = (_Float16)sc[ln][kt * 32 + kk];
    }
    v16h bv0, bv1;
#pragma unroll
    for (int j = 0; j < 16; ++j) {
      int key = kt * 32 + hi * 16 + j;
      bv0[j] = (_Float16)Vb[(size_t)key * ld + ln];
      bv1[j] = (_Float16)Vb[(size_t)key * ld + 16 + ln];
    }
    o0 = WMMA(ap, bv0, o0);
    o1 = WMMA(ap, bv1, o1);
  }
  float* Ob = O + (size_t)(b * 512 + t0) * ld + h * 32;
#pragma unroll
  for (int j = 0; j < 8; ++j) {
    int m = j + 8 * hi;
    Ob[(size_t)m * ld + ln]      = o0[j];
    Ob[(size_t)m * ld + 16 + ln] = o1[j];
  }
}

// ---------------------------------------------------------------------------
// Persistent bidirectional GRU scan. grid.x = 2 (fwd/bwd). Wh (128x384) f16
// lives in LDS for the whole scan; hidden state in LDS; WMMA per step.
// xg = x@Wx + bx precomputed: [b][t][384]. out: [b][t][256] (fwd|bwd halves).
// ---------------------------------------------------------------------------
#define GRU_OFF_WS 0
#define GRU_OFF_HS (384 * 128 * 2)
#define GRU_OFF_HF (GRU_OFF_HS + 16 * 128 * 2)
#define GRU_OFF_HG (GRU_OFF_HF + 8 * 128 * 4)
#define GRU_OFF_BH (GRU_OFF_HG + 8 * 384 * 4)
#define GRU_LDS_BYTES (GRU_OFF_BH + 384 * 4)

__global__ __launch_bounds__(256) void k_gru_scan(const float* __restrict__ xgf,
                                                  const float* __restrict__ xgb,
                                                  const float* __restrict__ Whf,
                                                  const float* __restrict__ Whb,
                                                  const float* __restrict__ bhf,
                                                  const float* __restrict__ bhb,
                                                  float* __restrict__ out, int out_ld) {
  extern __shared__ __align__(128) char smem[];
  _Float16* Ws  = (_Float16*)(smem + GRU_OFF_WS);   // [384 n][128 k]
  _Float16* Hs  = (_Float16*)(smem + GRU_OFF_HS);   // [16 m][128 k] (rows 8..15 zero)
  float*    hf  = (float*)(smem + GRU_OFF_HF);      // [8][128]
  float*    hg  = (float*)(smem + GRU_OFF_HG);      // [8][384]
  float*    bhs = (float*)(smem + GRU_OFF_BH);      // [384]

  const int dir  = blockIdx.x;
  const float* xg = dir ? xgb : xgf;
  const float* Wh = dir ? Whb : Whf;
  const float* bh = dir ? bhb : bhf;
  const int col0 = dir ? 128 : 0;

  const int tid = threadIdx.x, lane = tid & 31, wave = tid >> 5;
  for (int i = tid; i < 384 * 128; i += 256) {
    int n = i >> 7, k = i & 127;
    Ws[n * 128 + k] = (_Float16)Wh[(size_t)k * 384 + n];
  }
  for (int i = tid; i < 384; i += 256) bhs[i] = bh[i];
  for (int i = tid; i < 16 * 128; i += 256) Hs[i] = (_Float16)0.0f;
  for (int i = tid; i < 8 * 128; i += 256) hf[i] = 0.0f;
  __syncthreads();

  for (int t = 0; t < 512; ++t) {
    const int tt = dir ? (511 - t) : t;
    // hg = h @ Wh + bh  : 24 n-tiles, 3 per wave
    v8f acc[3]; acc[0] = v8f{}; acc[1] = v8f{}; acc[2] = v8f{};
#pragma unroll
    for (int kc = 0; kc < 4; ++kc) {
      v16h a = frag_a(Hs, 128, 0, kc * 32, lane);
#pragma unroll
      for (int nti = 0; nti < 3; ++nti) {
        v16h b = frag_b(Ws, 128, (wave * 3 + nti) * 16, kc * 32, lane);
        acc[nti] = WMMA(a, b, acc[nti]);
      }
    }
    {
      int hi = lane >> 4, ln = lane & 15;
#pragma unroll
      for (int nti = 0; nti < 3; ++nti) {
        int n = (wave * 3 + nti) * 16 + ln;
#pragma unroll
        for (int j = 0; j < 8; ++j) {
          int m = j + 8 * hi;
          if (m < 8) hg[m * 384 + n] = acc[nti][j] + bhs[n];
        }
      }
    }
    __syncthreads();
    // gates: 8 batches x 128 units
#pragma unroll
    for (int ii = 0; ii < 4; ++ii) {
      int e = tid + ii * 256;
      int bi = e >> 7, n = e & 127;
      size_t xb = (size_t)(bi * 512 + tt) * 384;
      float z = sigf(xg[xb + n]       + hg[bi * 384 + n]);
      float r = sigf(xg[xb + 128 + n] + hg[bi * 384 + 128 + n]);
      float hc = tanhf(xg[xb + 256 + n] + r * hg[bi * 384 + 256 + n]);
      float hn = z * hf[e] + (1.0f - z) * hc;
      hf[e] = hn;
      Hs[bi * 128 + n] = (_Float16)hn;
      out[(size_t)(bi * 512 + tt) * out_ld + col0 + n] = hn;
    }
    __syncthreads();
  }
}

// ---------------------------------------------------------------------------
// Persistent bidirectional LSTM scan (Wh 128x512 f16 in LDS = 128KB).
// xg = x@Wx + b precomputed [b][t][512]; out [b][t][256].
// ---------------------------------------------------------------------------
#define LSTM_OFF_WS 0
#define LSTM_OFF_HS (512 * 128 * 2)
#define LSTM_OFF_CS (LSTM_OFF_HS + 16 * 128 * 2)
#define LSTM_OFF_HG (LSTM_OFF_CS + 8 * 128 * 4)
#define LSTM_LDS_BYTES (LSTM_OFF_HG + 8 * 512 * 4)

__global__ __launch_bounds__(256) void k_lstm_scan(const float* __restrict__ xgf,
                                                   const float* __restrict__ xgb,
                                                   const float* __restrict__ Whf,
                                                   const float* __restrict__ Whb,
                                                   float* __restrict__ out, int out_ld) {
  extern __shared__ __align__(128) char smem[];
  _Float16* Ws = (_Float16*)(smem + LSTM_OFF_WS);   // [512 n][128 k]
  _Float16* Hs = (_Float16*)(smem + LSTM_OFF_HS);   // [16 m][128 k]
  float*    cs = (float*)(smem + LSTM_OFF_CS);      // [8][128]
  float*    hg = (float*)(smem + LSTM_OFF_HG);      // [8][512]

  const int dir  = blockIdx.x;
  const float* xg = dir ? xgb : xgf;
  const float* Wh = dir ? Whb : Whf;
  const int col0 = dir ? 128 : 0;

  const int tid = threadIdx.x, lane = tid & 31, wave = tid >> 5;
  for (int i = tid; i < 512 * 128; i += 256) {
    int n = i >> 7, k = i & 127;
    Ws[n * 128 + k] = (_Float16)Wh[(size_t)k * 512 + n];
  }
  for (int i = tid; i < 16 * 128; i += 256) Hs[i] = (_Float16)0.0f;
  for (int i = tid; i < 8 * 128; i += 256) cs[i] = 0.0f;
  __syncthreads();

  for (int t = 0; t < 512; ++t) {
    const int tt = dir ? (511 - t) : t;
    v8f acc[4]; acc[0] = v8f{}; acc[1] = v8f{}; acc[2] = v8f{}; acc[3] = v8f{};
#pragma unroll
    for (int kc = 0; kc < 4; ++kc) {
      v16h a = frag_a(Hs, 128, 0, kc * 32, lane);
#pragma unroll
      for (int nti = 0; nti < 4; ++nti) {
        v16h b = frag_b(Ws, 128, (wave * 4 + nti) * 16, kc * 32, lane);
        acc[nti] = WMMA(a, b, acc[nti]);
      }
    }
    {
      int hi = lane >> 4, ln = lane & 15;
#pragma unroll
      for (int nti = 0; nti < 4; ++nti) {
        int n = (wave * 4 + nti) * 16 + ln;
#pragma unroll
        for (int j = 0; j < 8; ++j) {
          int m = j + 8 * hi;
          if (m < 8) hg[m * 512 + n] = acc[nti][j];
        }
      }
    }
    __syncthreads();
#pragma unroll
    for (int ii = 0; ii < 4; ++ii) {
      int e = tid + ii * 256;
      int bi = e >> 7, n = e & 127;
      size_t xb = (size_t)(bi * 512 + tt) * 512;
      float gi = xg[xb + n]       + hg[bi * 512 + n];
      float gf = xg[xb + 128 + n] + hg[bi * 512 + 128 + n];
      float gc = xg[xb + 256 + n] + hg[bi * 512 + 256 + n];
      float go = xg[xb + 384 + n] + hg[bi * 512 + 384 + n];
      float c  = sigf(gf) * cs[e] + sigf(gi) * tanhf(gc);
      float hn = sigf(go) * tanhf(c);
      cs[e] = c;
      Hs[bi * 128 + n] = (_Float16)hn;
      out[(size_t)(bi * 512 + tt) * out_ld + col0 + n] = hn;
    }
    __syncthreads();
  }
}

// ===========================================================================
// Host side
// ===========================================================================
static inline void gemm(hipStream_t s, const float* A, int lda, long long sA,
                        const float* B, int ldb, long long sB, const float* bias,
                        float* C, int ldc, long long sC, int M, int N, int K, int act,
                        int batch = 1, int ksz = 0, int cin = 0, int pad = 0, int len = 512) {
  dim3 g((N + 63) / 64, (M + 63) / 64, batch);
  k_gemm<<<g, 256, 0, s>>>(A, lda, sA, B, ldb, sB, bias, C, ldc, sC, M, N, K, act,
                           ksz, cin, pad, len);
}
static inline void lnrm(hipStream_t s, const float* x, int ldx, float* y, int ldy,
                        const float* g, const float* b, int D) {
  k_ln<<<512, 256, 0, s>>>(x, ldx, y, ldy, g, b, 4096, D);
}
// cols must be a multiple of 4 (true for every concat width in this model)
static inline void copyc(hipStream_t s, const float* src, int lds_, float* dst, int ldd, int cols) {
  k_copy_async<<<2048, 256, 0, s>>>(src, lds_, dst, ldd, 4096, cols / 4);
}

struct CB {
  const float *convW, *convb, *denseW, *denseb, *invW, *invb, *ln1b, *ln1g, *ln2b, *ln2g;
};

// conv_block: y = LN2(Dense(GELU(Inv(LN1(Conv(x)))))) written into ydst slice.
static void conv_block_h(hipStream_t s, const float* x, int ldx, int cin, int h, int ksz, int pad,
                         const CB& p, float* ydst, int ldy, float* t1, float* t2) {
  gemm(s, x, ldx, 0, p.convW, h, 0, p.convb, t1, h, 0, 4096, h, ksz * cin, 0, 1, ksz, cin, pad, 512);
  lnrm(s, t1, h, t1, h, p.ln1g, p.ln1b, h);
  gemm(s, t1, h, 0, p.invW, 3 * h, 0, p.invb, t2, 3 * h, 0, 4096, 3 * h, h, 2 /*gelu*/);
  gemm(s, t2, 3 * h, 0, p.denseW, h, 0, p.denseb, t1, h, 0, 4096, h, 3 * h, 0);
  lnrm(s, t1, h, ydst, ldy, p.ln2g, p.ln2b, h);
}

extern "C" void kernel_launch(void* const* d_in, const int* in_sizes, int n_in,
                              void* d_out, int out_size, void* d_ws, size_t ws_size,
                              hipStream_t stream) {
  // ---- inputs: disambiguate node/edge order at runtime via sizes ----
  int iNode = 0, iEdge = 1;
  if (in_sizes[0] == 8 * 512 * 512 * 5) { iEdge = 0; iNode = 1; }
  const float* node_in = (const float*)d_in[iNode];
  const float* edge_in = (const float*)d_in[iEdge];

  // ---- params flattened in jax pytree order (dict keys sorted, lists in order)
#define PP(i) ((const float*)d_in[i])
  int q = 2;
  // attn: Wk, Wo, Wq, Wv, bk, bo, bq, bv
  const float* aWk = PP(q++); const float* aWo = PP(q++); const float* aWq = PP(q++);
  const float* aWv = PP(q++); const float* abk = PP(q++); const float* abo = PP(q++);
  const float* abq = PP(q++); const float* abv = PP(q++);
  auto readCB = [&]() {
    CB c;
    c.convW = PP(q++); c.convb = PP(q++);
    c.denseW = PP(q++); c.denseb = PP(q++);
    c.invW = PP(q++); c.invb = PP(q++);
    c.ln1b = PP(q++); c.ln1g = PP(q++);
    c.ln2b = PP(q++); c.ln2g = PP(q++);
    return c;
  };
  CB cb0 = readCB(), cb1 = readCB();           // params['cb'][0..1]
  CB cb15 = readCB(), cb30 = readCB(), cb6 = readCB();
  const float* edgeW = PP(q++);
  const float* ffnW[2]; const float* ffnb[2];
  ffnW[0] = PP(q++); ffnb[0] = PP(q++); ffnW[1] = PP(q++); ffnb[1] = PP(q++);
  const float* gnnW[2][6]; const float* gnnb[2][6];
  for (int i = 0; i < 2; ++i)
    for (int k = 0; k < 6; ++k) { gnnW[i][k] = PP(q++); gnnb[i][k] = PP(q++); }
  const float* ln1b_[6]; const float* ln1g_[6];
  for (int k = 0; k < 6; ++k) { ln1b_[k] = PP(q++); ln1g_[k] = PP(q++); }
  const float* ln2b = PP(q++); const float* ln2g = PP(q++);
  const float* preW = PP(q++);
  // rnn[0]: bwd{Wh,Wx,bh,bx}, fwd{Wh,Wx,bh,bx}
  const float* gWh_b = PP(q++); const float* gWx_b = PP(q++);
  const float* gbh_b = PP(q++); const float* gbx_b = PP(q++);
  const float* gWh_f = PP(q++); const float* gWx_f = PP(q++);
  const float* gbh_f = PP(q++); const float* gbx_f = PP(q++);
  // rnn[1]: bwd{Wh,Wx,b}, fwd{Wh,Wx,b}
  const float* lWh_b = PP(q++); const float* lWx_b = PP(q++); const float* lb_b = PP(q++);
  const float* lWh_f = PP(q++); const float* lWx_f = PP(q++); const float* lb_f = PP(q++);
#undef PP

  // ---- workspace carve ----
  size_t off = 0;
  auto alloc = [&](size_t floats) -> float* {
    float* p = (float*)((char*)d_ws + off);
    off += ((floats * 4 + 255) / 256) * 256;
    return p;
  };
  const size_t R = 4096;                       // B*N rows
  const size_t PLANE = 8ull * 512 * 512;
  float* saved  = alloc(R * 784);              // [node|cb6|cb15|cb30] concat, = 'saved'
  float* edges  = alloc(6 * PLANE);            // 6 adjacency planes
  float* cbout  = alloc(R * 912);              // conv_block output inside loop
  float* agg    = alloc(R * 912);              // GNN aggregation result
  float* extras = alloc(R * 1680);             // [cbout | 6x ex] concat
  float* node0b = alloc(R * 2464);             // [extras | saved]
  float* cat2   = alloc(R * 2592);             // [attn | node0]
  float* t1     = alloc(R * 128);
  float* t2     = alloc(R * 384);
  float* qb     = alloc(R * 64);
  float* kb     = alloc(R * 64);
  float* vb     = alloc(R * 64);
  float* ob     = alloc(R * 64);
  float* aproj  = alloc(R * 128);
  float* ffnout = alloc(R * 128);
  float* xgf    = alloc(R * 512);
  float* xgb    = alloc(R * 512);
  float* rnn0   = alloc(R * 256);
  (void)ws_size; (void)n_in; (void)out_size;

  hipStream_t s = stream;

  // ---- 1. pre: node = x[..,:14] @ preW  -> saved[:,0:128] ----
  gemm(s, node_in, 15, 0, preW, 128, 0, nullptr, saved, 784, 0, 4096, 128, 14, 0);

  // ---- 2. conv pyramid into 'saved' layout:
  // [0:128)=node [128:192)=y6 [192:320)=node [320:352)=y15 [352:544)=cb6_out
  // [544:560)=y30 [560:784)=cb15_out
  conv_block_h(s, saved, 784, 128, 64, 6, 2, cb6, saved + 128, 784, t1, t2);
  copyc(s, saved, 784, saved + 192, 784, 128);
  conv_block_h(s, saved + 128, 784, 192, 32, 15, 7, cb15, saved + 320, 784, t1, t2);
  copyc(s, saved + 128, 784, saved + 352, 784, 192);
  conv_block_h(s, saved + 320, 784, 224, 16, 30, 14, cb30, saved + 544, 784, t1, t2);
  copyc(s, saved + 320, 784, saved + 560, 784, 224);

  // ---- 3. edge planes ----
  k_edges<<<2048, 256, 0, s>>>(edge_in, edgeW, edges);

  const float* attn_b[3] = {abq, abk, abv};
  const float* attn_W[3] = {aWq, aWk, aWv};
  float* qkv[3] = {qb, kb, vb};

  // ---- 4. main loop ----
  const float* x_in = saved; int ld_in = 784, cin = 784;
  for (int i = 0; i < 2; ++i) {
    const int wcb = (i == 0) ? 912 : 384;
    const CB& cbp = (i == 0) ? cb0 : cb1;
    // conv_block -> cbout = [y(128) | x(cin)]
    conv_block_h(s, x_in, ld_in, cin, 128, 3, 1, cbp, cbout, wcb, t1, t2);
    copyc(s, x_in, ld_in, cbout + 128, wcb, cin);

    // GNN: extras = [cbout | ex_0..5]
    const int wex = wcb + 6 * 128;
    copyc(s, cbout, wcb, extras, wex, wcb);
    for (int k = 0; k < 6; ++k) {
      // agg[b] = adj_k[b](512x512) @ node0[b](512 x wcb)   (batched WMMA GEMM)
      gemm(s, edges + (size_t)k * PLANE, 512, 512ll * 512,
           cbout, wcb, 512ll * wcb, nullptr,
           agg, wcb, 512ll * wcb, 512, wcb, 512, 0, 8);
      gemm(s, agg, wcb, 0, gnnW[i][k], 128, 0, gnnb[i][k], t1, 128, 0, 4096, 128, wcb, 0);
      lnrm(s, t1, 128, extras + wcb + k * 128, wex, ln1g_[k], ln1b_[k], 128);
    }

    // node0 = [extras | saved]
    const int wn0 = wex + 784;
    copyc(s, extras, wex, node0b, wn0, wex);
    copyc(s, saved, 784, node0b + wex, wn0, 784);

    // ffn + LN2
    gemm(s, node0b, wn0, 0, ffnW[i], 128, 0, ffnb[i], t1, 128, 0, 4096, 128, wn0, 0);
    lnrm(s, t1, 128, ffnout, 128, ln2g, ln2b, 128);

    // MHA (shared weights across i)
    for (int j = 0; j < 3; ++j)
      gemm(s, ffnout, 128, 0, attn_W[j], 64, 0, attn_b[j], qkv[j], 64, 0, 4096, 64, 128, 0);
    k_attn<<<dim3(32, 16), 32, 0, s>>>(qb, kb, vb, ob);
    gemm(s, ob, 64, 0, aWo, 128, 0, abo, aproj, 128, 0, 4096, 128, 64, 0);

    // cat2 = [attn_out | node0]
    const int wc2 = 128 + wn0;
    copyc(s, aproj, 128, cat2, wc2, 128);
    copyc(s, node0b, wn0, cat2 + 128, wc2, wn0);

    if (i == 0) {
      // GRU: xg = cat2 @ Wx + bx per direction, then persistent scan
      gemm(s, cat2, wc2, 0, gWx_f, 384, 0, gbx_f, xgf, 384, 0, 4096, 384, wc2, 0);
      gemm(s, cat2, wc2, 0, gWx_b, 384, 0, gbx_b, xgb, 384, 0, 4096, 384, wc2, 0);
      k_gru_scan<<<2, 256, GRU_LDS_BYTES, s>>>(xgf, xgb, gWh_f, gWh_b, gbh_f, gbh_b, rnn0, 256);
      x_in = rnn0; ld_in = 256; cin = 256;
    } else {
      // LSTM -> final output
      gemm(s, cat2, wc2, 0, lWx_f, 512, 0, lb_f, xgf, 512, 0, 4096, 512, wc2, 0);
      gemm(s, cat2, wc2, 0, lWx_b, 512, 0, lb_b, xgb, 512, 0, 4096, 512, wc2, 0);
      k_lstm_scan<<<2, 256, LSTM_LDS_BYTES, s>>>(xgf, xgb, lWh_f, lWh_b, (float*)d_out, 256);
    }
  }
}
